// SSM_52338471469757
// MI455X (gfx1250) — matrix operand order repaired and checked by reading, not compiled, not run on hardware
//
#include <hip/hip_runtime.h>

typedef __bf16 bf16;
typedef float v8f  __attribute__((ext_vector_type(8)));
typedef bf16  v16bf __attribute__((ext_vector_type(16)));
typedef bf16  v8bf  __attribute__((ext_vector_type(8)));

union AFrag { v16bf v; v8bf h[2]; };

#define BATCH 8
#define HH    64
#define WW    64
#define LSEQ  4096            // HH*WW
#define DM    192             // D_MODEL
#define EI    384             // E_INNER
#define NST   16              // D_STATE
#define RDT   12              // DT_RANK
#define NXD   44              // DT_RANK + 2*D_STATE
#define ROWS  (BATCH * LSEQ)  // 32768

// ---------------------------------------------------------------- utilities
__global__ __launch_bounds__(256) void k_cvt_bf16(const float* __restrict__ in,
                                                  bf16* __restrict__ out, int n) {
  int i = blockIdx.x * 256 + threadIdx.x;
  if (i < n) out[i] = (bf16)in[i];
}

__device__ __forceinline__ float silu_f(float v) {
  return v / (1.f + __expf(-v));
}

// ---------------------------------------------------------------- GEMM 1
// xz = x @ W_in^T : A (32768 x 192) bf16 row-major, W (768 x 192) bf16 (N,K).
// One wave -> 16x64 strip. Split columns into xs (n<384) and z (n>=384).
__global__ __launch_bounds__(256) void k_gemm_in(const bf16* __restrict__ Abf,
                                                 const bf16* __restrict__ Wbf,
                                                 float* __restrict__ xs,
                                                 float* __restrict__ z) {
  const int K = DM;  // 192
  int wid    = (int)((blockIdx.x * 256 + threadIdx.x) >> 5);
  int lane   = threadIdx.x & 31;
  int mTile  = wid / 12;           // 2048 tiles of 16 rows
  int nGroup = wid - mTile * 12;   // 12 groups of 64 cols
  int mRow   = mTile << 4;
  int r = lane & 15, hi = lane >> 4;

  v8f acc[4];
#pragma unroll
  for (int j = 0; j < 4; ++j)
#pragma unroll
    for (int i = 0; i < 8; ++i) acc[j][i] = 0.f;

  const bf16* aRow = Abf + (size_t)(mRow + r) * K;
  int colBase = nGroup * 64 + r;

#pragma unroll
  for (int k0 = 0; k0 < K; k0 += 32) {
    AFrag a;
    const bf16* pa = aRow + k0 + hi * 16;
    a.h[0] = *(const v8bf*)(pa);
    a.h[1] = *(const v8bf*)(pa + 8);
    int kb = k0 + hi * 16;
    const bf16* pb = Wbf + (size_t)colBase * K + kb;
    v16bf b0 = *(const v16bf*)(pb);
    v16bf b1 = *(const v16bf*)(pb + (size_t)16 * K);
    v16bf b2 = *(const v16bf*)(pb + (size_t)32 * K);
    v16bf b3 = *(const v16bf*)(pb + (size_t)48 * K);
    acc[0] = __builtin_amdgcn_wmma_f32_16x16x32_bf16(false, a.v, false, b0, (short)0, acc[0], false, false);
    acc[1] = __builtin_amdgcn_wmma_f32_16x16x32_bf16(false, a.v, false, b1, (short)0, acc[1], false, false);
    acc[2] = __builtin_amdgcn_wmma_f32_16x16x32_bf16(false, a.v, false, b2, (short)0, acc[2], false, false);
    acc[3] = __builtin_amdgcn_wmma_f32_16x16x32_bf16(false, a.v, false, b3, (short)0, acc[3], false, false);
  }

#pragma unroll
  for (int j = 0; j < 4; ++j) {
    int n = nGroup * 64 + j * 16 + r;
#pragma unroll
    for (int v = 0; v < 8; ++v) {
      int m = mRow + v + hi * 8;
      float val = acc[j][v];
      if (n < EI) xs[(size_t)m * EI + n] = val;
      else        z [(size_t)m * EI + (n - EI)] = val;
    }
  }
}

// ---------------------------------------------------------------- GEMM out
// out = ycomb @ W_out^T : A (32768 x 384) bf16, W (192 x 384) bf16 (N,K).
__global__ __launch_bounds__(256) void k_gemm_out(const bf16* __restrict__ Abf,
                                                  const bf16* __restrict__ Wbf,
                                                  float* __restrict__ out) {
  const int K = EI;  // 384
  int wid    = (int)((blockIdx.x * 256 + threadIdx.x) >> 5);
  int lane   = threadIdx.x & 31;
  int mTile  = wid / 3;            // 2048 tiles of 16 rows
  int nGroup = wid - mTile * 3;    // 3 groups of 64 cols (N=192)
  int mRow   = mTile << 4;
  int r = lane & 15, hi = lane >> 4;

  v8f acc[4];
#pragma unroll
  for (int j = 0; j < 4; ++j)
#pragma unroll
    for (int i = 0; i < 8; ++i) acc[j][i] = 0.f;

  const bf16* aRow = Abf + (size_t)(mRow + r) * K;
  int colBase = nGroup * 64 + r;

  for (int k0 = 0; k0 < K; k0 += 32) {
    AFrag a;
    const bf16* pa = aRow + k0 + hi * 16;
    a.h[0] = *(const v8bf*)(pa);
    a.h[1] = *(const v8bf*)(pa + 8);
    int kb = k0 + hi * 16;
    const bf16* pb = Wbf + (size_t)colBase * K + kb;
    v16bf b0 = *(const v16bf*)(pb);
    v16bf b1 = *(const v16bf*)(pb + (size_t)16 * K);
    v16bf b2 = *(const v16bf*)(pb + (size_t)32 * K);
    v16bf b3 = *(const v16bf*)(pb + (size_t)48 * K);
    acc[0] = __builtin_amdgcn_wmma_f32_16x16x32_bf16(false, a.v, false, b0, (short)0, acc[0], false, false);
    acc[1] = __builtin_amdgcn_wmma_f32_16x16x32_bf16(false, a.v, false, b1, (short)0, acc[1], false, false);
    acc[2] = __builtin_amdgcn_wmma_f32_16x16x32_bf16(false, a.v, false, b2, (short)0, acc[2], false, false);
    acc[3] = __builtin_amdgcn_wmma_f32_16x16x32_bf16(false, a.v, false, b3, (short)0, acc[3], false, false);
  }

#pragma unroll
  for (int j = 0; j < 4; ++j) {
    int n = nGroup * 64 + j * 16 + r;
#pragma unroll
    for (int v = 0; v < 8; ++v) {
      int m = mRow + v + hi * 8;
      out[(size_t)m * DM + n] = acc[j][v];
    }
  }
}

// ---------------------------------------------------------------- depthwise 7x7 conv + SiLU
// xs viewed as (B, H, W, E) channel-last; coalesced along e.
__global__ __launch_bounds__(256) void k_conv(const float* __restrict__ xs,
                                              const float* __restrict__ cw,
                                              const float* __restrict__ cb,
                                              float* __restrict__ u) {
  size_t idx = (size_t)blockIdx.x * 256 + threadIdx.x;  // B*H*W*E
  int e = (int)(idx % EI);
  size_t s = idx / EI;
  int w  = (int)(s % WW);
  size_t s2 = s / WW;
  int h  = (int)(s2 % HH);
  int b  = (int)(s2 / HH);

  float acc = cb[e];
  const float* wp = cw + (size_t)e * 49;
#pragma unroll
  for (int di = 0; di < 7; ++di) {
    int hh = h + di - 3;
    if (hh < 0 || hh >= HH) continue;
#pragma unroll
    for (int dj = 0; dj < 7; ++dj) {
      int ww2 = w + dj - 3;
      if (ww2 < 0 || ww2 >= WW) continue;
      acc += wp[di * 7 + dj] * xs[(((size_t)b * HH + hh) * WW + ww2) * EI + e];
    }
  }
  u[idx] = silu_f(acc);
}

// ---------------------------------------------------------------- x_dbl = u @ W_x^T  (N=44)
__global__ __launch_bounds__(256) void k_xdbl(const float* __restrict__ u,
                                              const float* __restrict__ Wx,
                                              float* __restrict__ xdbl) {
  int idx = blockIdx.x * 256 + threadIdx.x;  // ROWS*44
  int c   = idx % NXD;
  int row = idx / NXD;
  const float4* ur = (const float4*)(u  + (size_t)row * EI);
  const float4* wr = (const float4*)(Wx + (size_t)c   * EI);
  float acc = 0.f;
#pragma unroll 4
  for (int k = 0; k < EI / 4; ++k) {
    float4 a = ur[k], ww = wr[k];
    acc += a.x * ww.x + a.y * ww.y + a.z * ww.z + a.w * ww.w;
  }
  xdbl[idx] = acc;
}

// ---------------------------------------------------------------- delta = softplus(dt_r@W_dt^T + 2*b_dt)
// (reference adds b_dt inside dt_lin AND again before softplus)
__global__ __launch_bounds__(256) void k_delta(const float* __restrict__ xdbl,
                                               const float* __restrict__ Wdt,
                                               const float* __restrict__ bdt,
                                               float* __restrict__ delta) {
  size_t idx = (size_t)blockIdx.x * 256 + threadIdx.x;  // ROWS*EI
  int e = (int)(idx % EI);
  size_t row = idx / EI;
  const float* xr = xdbl + row * NXD;
  const float* wr = Wdt + (size_t)e * RDT;
  float acc = 2.f * bdt[e];
#pragma unroll
  for (int k = 0; k < RDT; ++k) acc += xr[k] * wr[k];
  delta[idx] = (acc > 20.f) ? acc : log1pf(__expf(acc));
}

// ---------------------------------------------------------------- selective scan
// wave32 = 2 channels x 16 states (lane&15 == state index n).
__global__ __launch_bounds__(256) void k_scan(const float* __restrict__ delta,
                                              const float* __restrict__ u,
                                              const float* __restrict__ xdbl,
                                              const float* __restrict__ A_log,
                                              const float* __restrict__ Dp,
                                              float* __restrict__ ys) {
  int wid  = (int)((blockIdx.x * 256 + threadIdx.x) >> 5);
  int lane = threadIdx.x & 31;
  int half = lane >> 4;
  int n    = lane & 15;
  int ch   = wid * 2 + half;   // 0..3071
  int b    = ch / EI;
  int e    = ch - b * EI;

  float Aneg = -__expf(A_log[e * NST + n]);
  float dpe  = Dp[e];
  float state = 0.f;

  const float* dPtr = delta + (size_t)b * LSEQ * EI + e;
  const float* uPtr = u     + (size_t)b * LSEQ * EI + e;
  const float* xPtr = xdbl  + (size_t)b * LSEQ * NXD;
  float* yPtr       = ys    + (size_t)b * LSEQ * EI + e;

  for (int t = 0; t < LSEQ; ++t) {
    float dv = dPtr[(size_t)t * EI];
    float uv = uPtr[(size_t)t * EI];
    const float* xr = xPtr + (size_t)t * NXD;
    float Bv = xr[RDT + n];
    float Cv = xr[RDT + NST + n];
    // pull upcoming timesteps toward the WGP (global_prefetch_b8)
    __builtin_prefetch(dPtr + (size_t)(t + 24) * EI, 0, 1);
    __builtin_prefetch(xr + (size_t)24 * NXD, 0, 1);

    float dA = __expf(dv * Aneg);
    state = state * dA + (dv * uv) * Bv;
    float y = state * Cv;
    y += __shfl_xor(y, 1, 16);
    y += __shfl_xor(y, 2, 16);
    y += __shfl_xor(y, 4, 16);
    y += __shfl_xor(y, 8, 16);
    if (n == 0) yPtr[(size_t)t * EI] = y + uv * dpe;
  }
}

// ---------------------------------------------------------------- batch sum + combine
__global__ __launch_bounds__(256) void k_bsum(const float* __restrict__ ys,
                                              float* __restrict__ ysum) {
  int idx = blockIdx.x * 256 + threadIdx.x;  // LSEQ*EI
  float s = 0.f;
#pragma unroll
  for (int b = 0; b < BATCH; ++b) s += ys[(size_t)b * LSEQ * EI + idx];
  ysum[idx] = s;
}

__global__ __launch_bounds__(256) void k_combine(const float* __restrict__ ysum,
                                                 const float* __restrict__ z,
                                                 bf16* __restrict__ yc) {
  size_t idx = (size_t)blockIdx.x * 256 + threadIdx.x;  // B*L*EI
  size_t le = idx % ((size_t)LSEQ * EI);
  float zv = z[idx];
  yc[idx] = (bf16)(ysum[le] * silu_f(zv));
}

// ---------------------------------------------------------------- launch
extern "C" void kernel_launch(void* const* d_in, const int* in_sizes, int n_in,
                              void* d_out, int out_size, void* d_ws, size_t ws_size,
                              hipStream_t stream) {
  const float* x     = (const float*)d_in[0];
  // d_in[1], d_in[2] are H, W scalars (compile-time constants here)
  const float* W_in  = (const float*)d_in[3];
  const float* convw = (const float*)d_in[4];
  const float* convb = (const float*)d_in[5];
  const float* W_x   = (const float*)d_in[6];
  const float* W_dt  = (const float*)d_in[7];
  const float* b_dt  = (const float*)d_in[8];
  const float* A_log = (const float*)d_in[9];
  const float* Dp    = (const float*)d_in[10];
  const float* W_out = (const float*)d_in[11];
  float* out = (float*)d_out;

  const size_t BLE = (size_t)ROWS * EI;  // 12,582,912

  char* w = (char*)d_ws;
  size_t off = 0;
  auto wsAlloc = [&](size_t bytes) -> void* {
    void* p = (void*)(w + off);
    off = (off + bytes + 255) & ~(size_t)255;
    return p;
  };

  float* xs_buf   = (float*)wsAlloc(BLE * 4);              // xs; reused as ys after conv
  float* z_buf    = (float*)wsAlloc(BLE * 4);
  float* u_buf    = (float*)wsAlloc(BLE * 4);              // u; reused as ycomb (bf16) after scan
  float* dlt_buf  = (float*)wsAlloc(BLE * 4);
  float* xdbl_buf = (float*)wsAlloc((size_t)ROWS * NXD * 4);
  float* ysum_buf = (float*)wsAlloc((size_t)LSEQ * EI * 4);
  bf16*  x_bf     = (bf16*)wsAlloc((size_t)ROWS * DM * 2);
  bf16*  win_bf   = (bf16*)wsAlloc((size_t)2 * EI * DM * 2);
  bf16*  wout_bf  = (bf16*)wsAlloc((size_t)DM * EI * 2);
  float* ys_buf   = xs_buf;          // alias (xs dead after conv)
  bf16*  yc_buf   = (bf16*)u_buf;    // alias (u dead after scan)

  // 1) fp32 -> bf16 conversions for WMMA operands
  k_cvt_bf16<<<(ROWS * DM + 255) / 256, 256, 0, stream>>>(x, x_bf, ROWS * DM);
  k_cvt_bf16<<<(2 * EI * DM + 255) / 256, 256, 0, stream>>>(W_in, win_bf, 2 * EI * DM);
  k_cvt_bf16<<<(DM * EI + 255) / 256, 256, 0, stream>>>(W_out, wout_bf, DM * EI);

  // 2) input projection GEMM (WMMA bf16): 2048 mTiles * 12 nGroups = 24576 waves
  k_gemm_in<<<3072, 256, 0, stream>>>(x_bf, win_bf, xs_buf, z_buf);

  // 3) depthwise conv + SiLU
  k_conv<<<(int)(BLE / 256), 256, 0, stream>>>(xs_buf, convw, convb, u_buf);

  // 4) x_dbl projection (N=44, VALU dot)
  k_xdbl<<<(ROWS * NXD) / 256, 256, 0, stream>>>(u_buf, W_x, xdbl_buf);

  // 5) delta
  k_delta<<<(int)(BLE / 256), 256, 0, stream>>>(xdbl_buf, W_dt, b_dt, dlt_buf);

  // 6) selective scan: 3072 channels, 2 per wave, 8 waves/block -> 192 blocks
  k_scan<<<192, 256, 0, stream>>>(dlt_buf, u_buf, xdbl_buf, A_log, Dp, ys_buf);

  // 7) sum over batch, gate with silu(z), pack bf16
  k_bsum<<<(LSEQ * EI) / 256, 256, 0, stream>>>(ys_buf, ysum_buf);
  k_combine<<<(int)(BLE / 256), 256, 0, stream>>>(ysum_buf, z_buf, yc_buf);

  // 8) output projection GEMM (WMMA bf16): 2048 mTiles * 3 nGroups = 6144 waves
  k_gemm_out<<<768, 256, 0, stream>>>(yc_buf, wout_bf, out);

  (void)in_sizes; (void)n_in; (void)out_size; (void)ws_size;
}
